// ScaledDotProductAttention_50543175139751
// MI455X (gfx1250) — matrix operand order repaired and checked by reading, not compiled, not run on hardware
//
#include <hip/hip_runtime.h>
#include <hip/hip_bf16.h>

// Problem constants (from reference): B=8, D=512, N=512, L=512, DV=64
#define PB   8
#define PD   512
#define PN   512
#define PL   512
#define PDV  64
#define PNL  (PN * PL)          // 262144

typedef __attribute__((ext_vector_type(16))) __bf16 v16bf;
typedef __attribute__((ext_vector_type(8)))  float  v8f;

// ---- order-preserving float<->uint encoding for deterministic atomicMax ----
__device__ __forceinline__ unsigned ordEnc(float f) {
    unsigned u = __float_as_uint(f);
    return (u & 0x80000000u) ? ~u : (u | 0x80000000u);
}
__device__ __forceinline__ float ordDec(unsigned u) {
    return __uint_as_float((u & 0x80000000u) ? (u ^ 0x80000000u) : ~u);
}

// ---- wave32 butterfly reductions via ds_swizzle (group-of-32 xor mode) ----
// pattern = and_mask(0x1f) | or_mask(0)<<5 | xor_mask<<10
#define DS_SWZ_XOR(x, patt) \
    __int_as_float(__builtin_amdgcn_ds_swizzle(__float_as_int(x), (patt)))

__device__ __forceinline__ float waveRedSum(float x) {
    x += DS_SWZ_XOR(x, 0x041f);   // xor 1
    x += DS_SWZ_XOR(x, 0x081f);   // xor 2
    x += DS_SWZ_XOR(x, 0x101f);   // xor 4
    x += DS_SWZ_XOR(x, 0x201f);   // xor 8
    x += DS_SWZ_XOR(x, 0x401f);   // xor 16
    return x;
}
__device__ __forceinline__ float waveRedMax(float x) {
    x = fmaxf(x, DS_SWZ_XOR(x, 0x041f));
    x = fmaxf(x, DS_SWZ_XOR(x, 0x081f));
    x = fmaxf(x, DS_SWZ_XOR(x, 0x101f));
    x = fmaxf(x, DS_SWZ_XOR(x, 0x201f));
    x = fmaxf(x, DS_SWZ_XOR(x, 0x401f));
    return x;
}

__device__ __forceinline__ unsigned pack_bf16(float a, float b) {
    union { unsigned u; __bf16 h[2]; } x;
    x.h[0] = (__bf16)a; x.h[1] = (__bf16)b;
    return x.u;
}

// Compute one 16(n) x 16(l per wave) att tile accumulator with WMMA bf16.
// Block = 128 threads = 4 waves; wave w owns columns l0+16w .. l0+16w+15.
// Tiles are staged into LDS in WMMA *fragment order* (inverse of ISA 7.12.2
// per-lane layout), so each lane fetches its whole 16-elem fragment with two
// ds_load_b128's instead of 16 scalar ds_load_u16's.
//
// A-fragment layout (16x32 bf16): lane = ((K&8)?16:0)+M, elem i = (K&7)+((K>=16)?8:0)
// B-fragment layout (32x16 bf16): lane = ((K&8)?16:0)+Nl, elem i = (K&7)+((K>=16)?8:0)
__device__ __forceinline__ v8f attTileWmma(const float* __restrict__ kb,
                                           const float* __restrict__ qb,
                                           int n0, int l0, int tid,
                                           unsigned* aF,   // [32 lanes][8 dw]
                                           unsigned* bF) { // [4 waves][32 lanes][8 dw]
    const int lane = tid & 31;
    const int wave = tid >> 5;

    // staging roles (constant per thread)
    const int an = tid & 15;          // A: n within tile
    const int ah = (tid >> 4) & 1;    // A: K half (0: K<16, 1: K>=16)
    const int ag = tid >> 5;          // A: pair group 0..3
    const int bc = tid & 63;          // B: column within 64
    const int bh = tid >> 6;          // B: K half

    v8f c;
    #pragma unroll
    for (int i = 0; i < 8; ++i) c[i] = 0.0f;

    for (int d0 = 0; d0 < PD; d0 += 32) {
        __syncthreads();
        // ---- stage A (k tile, 16n x 32K): 2 packed b32 stores / thread ----
        #pragma unroll
        for (int p = 0; p < 2; ++p) {
            const int j = ag * 2 + p;              // 0..7
            const int K = ah * 16 + 2 * j;         // even
            const float f0 = kb[(size_t)(d0 + K    ) * PN + (n0 + an)];
            const float f1 = kb[(size_t)(d0 + K + 1) * PN + (n0 + an)];
            const int lA = ((K & 8) ? 16 : 0) + an;
            const int iA = (K & 7) + ((K >= 16) ? 8 : 0);   // even
            aF[lA * 8 + (iA >> 1)] = pack_bf16(f0, f1);
        }
        // ---- stage B (q tile, 32K x 64l): 8 packed b32 stores / thread ----
        #pragma unroll
        for (int j = 0; j < 8; ++j) {
            const int K = bh * 16 + 2 * j;         // even
            const float f0 = qb[(size_t)(d0 + K    ) * PL + (l0 + bc)];
            const float f1 = qb[(size_t)(d0 + K + 1) * PL + (l0 + bc)];
            const int w  = bc >> 4;
            const int lB = ((K & 8) ? 16 : 0) + (bc & 15);
            const int iB = (K & 7) + ((K >= 16) ? 8 : 0);   // even
            bF[(w * 32 + lB) * 8 + (iB >> 1)] = pack_bf16(f0, f1);
        }
        __syncthreads();

        const v16bf av = *(const v16bf*)(aF + lane * 8);
        const v16bf bv = *(const v16bf*)(bF + (wave * 32 + lane) * 8);
        c = __builtin_amdgcn_wmma_f32_16x16x32_bf16(
                false, av, false, bv, (short)0, c, false, false);
    }
    return c;
}

// ---------------- pass 0: init per-batch max ----------------
__global__ void initWsKernel(unsigned* maxOrd) {
    if (threadIdx.x < PB) maxOrd[threadIdx.x] = 0u;   // below ordEnc of any real
}

// ---------------- pass A: global max of logits per batch ----------------
__global__ void __launch_bounds__(128)
passA(const float* __restrict__ k, const float* __restrict__ q,
      unsigned* __restrict__ maxOrd) {
    __shared__ alignas(32) unsigned aF[32 * 8];
    __shared__ alignas(32) unsigned bF[4 * 32 * 8];

    const int bid = blockIdx.x;          // 2048 blocks
    const int b   = bid >> 8;            // 256 tiles per batch
    const int rem = bid & 255;
    const int n0  = (rem >> 3) << 4;     // 32 n-tiles of 16
    const int l0  = (rem & 7) << 6;      // 8 l-tiles of 64
    const int tid = threadIdx.x;

    const float* kb = k + (size_t)b * PD * PN;
    const float* qb = q + (size_t)b * PD * PL;

    v8f c = attTileWmma(kb, qb, n0, l0, tid, aF, bF);

    float vmax = c[0];
    #pragma unroll
    for (int r = 1; r < 8; ++r) vmax = fmaxf(vmax, c[r]);
    vmax = waveRedMax(vmax);
    if ((tid & 31) == 0) atomicMax(&maxOrd[b], ordEnc(vmax));
}

// ---------------- pass B: exp weights + v-weighted partial sums ----------------
__global__ void __launch_bounds__(128)
passB(const float* __restrict__ k, const float* __restrict__ q,
      const float* __restrict__ v, const unsigned* __restrict__ maxOrd,
      float* __restrict__ pAcc, float* __restrict__ pZ) {
    __shared__ alignas(32) unsigned aF[32 * 8];
    __shared__ alignas(32) unsigned bF[4 * 32 * 8];
    __shared__ float  wLds[16 * 64];
    __shared__ float  zWave[4];

    const int bid = blockIdx.x;
    const int b   = bid >> 8;
    const int rem = bid & 255;
    const int n0  = (rem >> 3) << 4;
    const int l0  = (rem & 7) << 6;
    const int tid = threadIdx.x;
    const int lane = tid & 31;
    const int wave = tid >> 5;

    const float* kb = k + (size_t)b * PD * PN;
    const float* qb = q + (size_t)b * PD * PL;

    v8f c = attTileWmma(kb, qb, n0, l0, tid, aF, bF);

    const float mmax = ordDec(maxOrd[b]);

    // C layout (ISA 7.12.2): lanes 0-15: N=lane, M=r; lanes 16-31: N=lane-16, M=8+r
    const int col   = (wave << 4) + (lane & 15);
    const int mrow0 = (lane >= 16) ? 8 : 0;
    float zpart = 0.0f;
    #pragma unroll
    for (int r = 0; r < 8; ++r) {
        float w = __expf(c[r] - mmax);
        wLds[(mrow0 + r) * 64 + col] = w;
        zpart += w;
    }
    zpart = waveRedSum(zpart);
    if (lane == 0) zWave[wave] = zpart;
    __syncthreads();

    // phase 2: stream v exactly once device-wide.
    // wave w handles dv in [16w, 16w+16). e-chunk outer: each w value is
    // loaded from LDS once and feeds 16 independent FMAs (16 v streams whose
    // j*PNL offsets fold into the VMEM instruction immediates).
    const float* vb = v + (size_t)b * PDV * PNL + (size_t)(wave << 4) * PNL;
    float accs[16];
    #pragma unroll
    for (int j = 0; j < 16; ++j) accs[j] = 0.0f;

    for (int e0 = 0; e0 < 1024; e0 += 32) {
        const int e  = e0 + lane;
        const float w = wLds[e];
        const int mm = e >> 6, ll = e & 63;
        const float* vp = vb + (size_t)(n0 + mm) * PL + (l0 + ll);
        __builtin_prefetch(vp + 2 * PL, 0, 0);   // next tile rows
        #pragma unroll
        for (int j = 0; j < 16; ++j)
            accs[j] = fmaf(w, vp[(size_t)j * PNL], accs[j]);
    }
    #pragma unroll
    for (int j = 0; j < 16; ++j) {
        float red = waveRedSum(accs[j]);
        if (lane == 0) pAcc[(size_t)bid * PDV + (wave << 4) + j] = red;
    }
    if (tid == 0) pZ[bid] = zWave[0] + zWave[1] + zWave[2] + zWave[3];
}

// ---------------- pass C: deterministic fixed-order reduction ----------------
__global__ void passC(const float* __restrict__ pAcc, const float* __restrict__ pZ,
                      float* __restrict__ out) {
    const int b  = blockIdx.x;     // 8
    const int dv = threadIdx.x;    // 64
    float acc = 0.0f;
    for (int t = 0; t < 256; ++t)
        acc += pAcc[(size_t)(b * 256 + t) * PDV + dv];
    float Z = 0.0f;
    for (int t = 0; t < 256; ++t)
        Z += pZ[b * 256 + t];
    out[b * PDV + dv] = acc / Z;
}

extern "C" void kernel_launch(void* const* d_in, const int* in_sizes, int n_in,
                              void* d_out, int out_size, void* d_ws, size_t ws_size,
                              hipStream_t stream) {
    const float* k = (const float*)d_in[0];   // (B, D, N)
    const float* q = (const float*)d_in[1];   // (B, D, L)
    const float* v = (const float*)d_in[2];   // (B, DV, N*L)
    float* out = (float*)d_out;               // (B, DV, 1) -> 512 floats

    // ws layout: [0..32)      unsigned maxOrd[8]
    //            [256..)      float pAcc[2048][64]
    //            after pAcc   float pZ[2048]
    unsigned* maxOrd = (unsigned*)d_ws;
    float* pAcc = (float*)((char*)d_ws + 256);
    float* pZ   = (float*)((char*)d_ws + 256 + (size_t)2048 * PDV * sizeof(float));
    (void)in_sizes; (void)n_in; (void)out_size; (void)ws_size;

    initWsKernel<<<1, 32, 0, stream>>>(maxOrd);
    passA<<<2048, 128, 0, stream>>>(k, q, maxOrd);
    passB<<<2048, 128, 0, stream>>>(k, q, v, maxOrd, pAcc, pZ);
    passC<<<8, PDV, 0, stream>>>(pAcc, pZ, out);
}
